// RNN_14078902797083
// MI455X (gfx1250) — matrix operand order repaired and checked by reading, not compiled, not run on hardware
//
#include <hip/hip_runtime.h>
#include <hip/hip_bf16.h>

// ---------------------------------------------------------------------------
// 2-layer tanh RNN on MI455X (gfx1250, wave32, WMMA bf16 -> f32).
//   phase 1: pack weights -> bf16 WMMA-B fragments; bias sums; x -> bf16 [T][B][256]
//   phase 2: input projection GEMM (templated K -> immediate-offset fragment loads)
//   phase 3: persistent single-workgroup recurrent scan; h double-buffered in LDS;
//            xw_t register-double-buffered so each step's first WMMA never waits
//            on a fresh global latency (critical: 4096 sequential steps)
//   phase 4: layer-1 projection + scan (final h only), FC + softmax head
// ---------------------------------------------------------------------------

typedef __attribute__((ext_vector_type(16))) __bf16 v16bf;
typedef __attribute__((ext_vector_type(8)))  __bf16 v8bf;
typedef __attribute__((ext_vector_type(8)))  float  v8f;

#define NB   64      // batch
#define NT   2048    // timesteps
#define NIN  256     // input size
#define NH   512     // hidden

#define HSTR 520     // LDS row stride in bf16 elems (512 + 8 pad -> bank spread)

// ---- workspace layout (bytes) ----------------------------------------------
#define OFF_XBF   ((size_t)0)                       // [T][B][256] bf16   64 MB
#define OFF_XW    ((size_t)67108864)                // [T][B][512] f32   256 MB (shared by both layers)
#define OFF_H1    (OFF_XW  + (size_t)268435456)     // [T][B][512] bf16  128 MB
#define OFF_WIH0  (OFF_H1  + (size_t)134217728)     // packed bf16 frags 256 KB
#define OFF_WHH0  (OFF_WIH0 + (size_t)262144)       // 512 KB
#define OFF_WIH1  (OFF_WHH0 + (size_t)524288)       // 512 KB
#define OFF_WHH1  (OFF_WIH1 + (size_t)524288)       // 512 KB
#define OFF_B0    (OFF_WHH1 + (size_t)524288)       // 512 f32
#define OFF_B1    (OFF_B0   + (size_t)2048)
#define OFF_HL    (OFF_B1   + (size_t)2048)         // [64][512] f32 final hidden

// ---------------------------------------------------------------------------
// x [B][T][256] f32  ->  xbf [T][B][256] bf16  (transpose so GEMM row = t*64+b)
// ---------------------------------------------------------------------------
__global__ void cvt_x_kernel(const float* __restrict__ x, __bf16* __restrict__ xbf) {
    size_t i = (size_t)blockIdx.x * 256 + threadIdx.x;   // over T*B*256 elems
    int k = (int)(i & 255);
    int b = (int)((i >> 8) & 63);
    int t = (int)(i >> 14);
    xbf[i] = (__bf16)x[((size_t)b * NT + t) * NIN + k];
}

// ---------------------------------------------------------------------------
// Pack W[n][k] (f32) into WMMA-B bf16 fragments, B[k][n]=W[n][k].
// Fragment (nt,kb): 32 lanes x 16 bf16 contiguous per lane.
//   lane l, elem j  <->  k = kb*32 + (j>>3)*16 + (l>>4)*8 + (j&7),  n = nt*16 + (l&15)
// (dense 16-bit B layout: lanes 0-15 hold K=0..7,16..23, lanes 16-31 hold K=8..15,24..31)
// ---------------------------------------------------------------------------
__global__ void pack_w_kernel(const float* __restrict__ W, __bf16* __restrict__ out, int K) {
    int KB = K >> 5;
    size_t p = (size_t)blockIdx.x * 256 + threadIdx.x;
    if (p >= (size_t)NH * K) return;
    int j  = (int)(p & 15);
    int l  = (int)((p >> 4) & 31);
    size_t f = p >> 9;
    int kb = (int)(f % KB);
    int nt = (int)(f / KB);
    int n = nt * 16 + (l & 15);
    int k = kb * 32 + (j >> 3) * 16 + (l >> 4) * 8 + (j & 7);
    out[p] = (__bf16)W[(size_t)n * K + k];
}

__global__ void add_bias_kernel(const float* __restrict__ a, const float* __restrict__ b,
                                float* __restrict__ o) {
    int i = blockIdx.x * 256 + threadIdx.x;
    if (i < NH) o[i] = a[i] + b[i];
}

// ---------------------------------------------------------------------------
// out[M][512] = A[M][K](bf16) x Bpacked + bias.  K is a template constant so
// every fragment load is base + immediate offset (no per-fragment 64-bit math).
// 256 threads = 8 waves; wave = 16x128 tile (1 M-tile x 8 N-tiles).
// ---------------------------------------------------------------------------
template<int K>
__global__ __launch_bounds__(256)
void gemm_bias_kernel(const __bf16* __restrict__ A, const __bf16* __restrict__ Bp,
                      const float* __restrict__ bias, float* __restrict__ out, int M) {
    constexpr int KB = K / 32;
    const int tid  = threadIdx.x;
    const int lane = tid & 31, wave = tid >> 5;
    const int g = lane >> 4, ln = lane & 15;
    const int mt_local = wave >> 2;         // 0..1
    const int ngroup   = wave & 3;          // covers n-tiles ngroup*8 .. +7
    const size_t mbase = ((size_t)blockIdx.x * 2 + mt_local) * 16;

    v8f acc[8];
#pragma unroll
    for (int nt = 0; nt < 8; ++nt) {
        float bv = bias[(ngroup * 8 + nt) * 16 + ln];
#pragma unroll
        for (int i = 0; i < 8; ++i) acc[nt][i] = bv;
    }

    // lane-dependent terms folded into bases once; loop offsets are immediates
    const __bf16* abase = A + (mbase + ln) * (size_t)K + g * 8;
    const __bf16* bbase = Bp + (size_t)(ngroup * 8) * KB * 512 + lane * 16;

#pragma unroll 2
    for (int kb = 0; kb < KB; ++kb) {
        union { v16bf v; v8bf h[2]; } a;
        a.h[0] = *(const v8bf*)(abase + kb * 32);
        a.h[1] = *(const v8bf*)(abase + kb * 32 + 16);
        const __bf16* bk = bbase + kb * 512;
#pragma unroll
        for (int nt = 0; nt < 8; ++nt) {
            v16bf bf = *(const v16bf*)(bk + nt * (KB * 512));   // imm offset
            acc[nt] = __builtin_amdgcn_wmma_f32_16x16x32_bf16(
                false, a.v, false, bf, (short)0, acc[nt], false, false);
        }
    }

    float* obase = out + (mbase + g * 8) * NH + ngroup * 128 + ln;
#pragma unroll
    for (int nt = 0; nt < 8; ++nt)
#pragma unroll
        for (int i = 0; i < 8; ++i)
            obase[(size_t)i * NH + nt * 16] = acc[nt][i];       // C: M = i + 8g
}

// ---------------------------------------------------------------------------
// Recurrent scan: h_t = tanh(xw_t + h_{t-1} @ W_hh^T).
// One persistent workgroup (32 waves, one WGP). h double-buffered in LDS
// ([2][64][520] bf16 = 133 KB of 320 KB); one barrier per step.
// xw_t is register double-buffered: step t seeds acc from regs loaded at t-1,
// then immediately issues loads for t+1 which complete under the 64-WMMA K-loop
// -> no global-latency stall at the head of any of the 2048 sequential steps.
// ---------------------------------------------------------------------------
__global__ __launch_bounds__(1024)
void rnn_scan_kernel(const float* __restrict__ xw, const __bf16* __restrict__ Wp,
                     __bf16* __restrict__ hout, float* __restrict__ hlast, int T) {
    extern __shared__ __bf16 lds[];                  // [2][64][HSTR]
    const int tid  = threadIdx.x;
    const int lane = tid & 31, wave = tid >> 5;
    const int g = lane >> 4, ln = lane & 15;
    const int ncol = wave * 16 + ln;                 // this lane's output column

    for (int i = tid; i < NB * HSTR; i += 1024) lds[i] = (__bf16)0.f;  // h_0 = 0

    // this wave's W_hh fragments: n-tile == wave, KB == 16, 512 elems per frag
    const __bf16* wbase = Wp + (size_t)wave * 16 * 512 + lane * 16;

    // prime the xw register buffer with step 0 (overlaps the LDS zero-fill)
    float xwreg[32];
    {
        const float* x0 = xw + (size_t)g * 8 * NH + ncol;
#pragma unroll
        for (int mt = 0; mt < 4; ++mt)
#pragma unroll
            for (int i = 0; i < 8; ++i)
                xwreg[mt * 8 + i] = x0[(mt * 16 + i) * NH];
    }
    __syncthreads();

    for (int t = 0; t < T; ++t) {
        const __bf16* rbuf = lds + (size_t)(t & 1) * (NB * HSTR);
        __bf16*       wbuf = lds + (size_t)((t & 1) ^ 1) * (NB * HSTR);

        // seed C with staged xw_t (bias folded in at projection time)
        v8f acc[4];
#pragma unroll
        for (int mt = 0; mt < 4; ++mt)
#pragma unroll
            for (int i = 0; i < 8; ++i)
                acc[mt][i] = xwreg[mt * 8 + i];

        // issue next step's xw loads now; consumed only at t+1 (t==T-1: reload)
        {
            int tn = (t + 1 < T) ? (t + 1) : t;
            const float* xn = xw + (size_t)tn * (NB * NH) + (size_t)g * 8 * NH + ncol;
            __builtin_prefetch(xn + NB * NH, 0, 0);          // speculative t+2 line
#pragma unroll
            for (int mt = 0; mt < 4; ++mt)
#pragma unroll
                for (int i = 0; i < 8; ++i)
                    xwreg[mt * 8 + i] = xn[(mt * 16 + i) * NH];
        }

        // K loop: A fragments from LDS (imm DS offsets), W fragments from L2
        const __bf16* abase = rbuf + (size_t)ln * HSTR + g * 8;
#pragma unroll 4
        for (int kb = 0; kb < 16; ++kb) {
            v16bf bf = *(const v16bf*)(wbase + (size_t)kb * 512);
#pragma unroll
            for (int mt = 0; mt < 4; ++mt) {
                union { v16bf v; v8bf h[2]; } a;
                const __bf16* ap = abase + mt * (16 * HSTR) + kb * 32;  // imm
                a.h[0] = *(const v8bf*)(ap);
                a.h[1] = *(const v8bf*)(ap + 16);
                acc[mt] = __builtin_amdgcn_wmma_f32_16x16x32_bf16(
                    false, a.v, false, bf, (short)0, acc[mt], false, false);
            }
        }

        // tanh, stage h_t for step t+1, optionally emit h_t / final h
        __bf16* wb = wbuf + (size_t)g * 8 * HSTR + ncol;
        __bf16* hb = hout ? hout + ((size_t)t * NB + g * 8) * NH + ncol : nullptr;
#pragma unroll
        for (int mt = 0; mt < 4; ++mt) {
#pragma unroll
            for (int i = 0; i < 8; ++i) {
                float v = acc[mt][i];
                float e = __expf(2.f * v);           // tanh(v) = 1 - 2/(e^{2v}+1)
                float th = 1.f - 2.f / (e + 1.f);
                __bf16 hv = (__bf16)th;
                wb[(mt * 16 + i) * HSTR] = hv;
                if (hb) hb[(size_t)(mt * 16 + i) * NH] = hv;
                if (hlast && t == T - 1)
                    hlast[(size_t)(mt * 16 + g * 8 + i) * NH + ncol] = th;
            }
        }
        __syncthreads();                             // h_t visible; WAR safe (double buffer)
    }
}

// ---------------------------------------------------------------------------
// logits = h_last @ fc_w^T + fc_b ; softmax over 2 classes (64x2, trivial).
// ---------------------------------------------------------------------------
__global__ void head_kernel(const float* __restrict__ h, const float* __restrict__ fcw,
                            const float* __restrict__ fcb, float* __restrict__ out) {
    int b = threadIdx.x;
    if (b >= NB) return;
    float d0 = fcb[0], d1 = fcb[1];
    for (int k = 0; k < NH; ++k) {
        float hv = h[b * NH + k];
        d0 += hv * fcw[k];
        d1 += hv * fcw[NH + k];
    }
    float m = fmaxf(d0, d1);
    float e0 = __expf(d0 - m), e1 = __expf(d1 - m);
    float s = e0 + e1;
    out[b * 2 + 0] = e0 / s;
    out[b * 2 + 1] = e1 / s;
}

// ---------------------------------------------------------------------------
extern "C" void kernel_launch(void* const* d_in, const int* in_sizes, int n_in,
                              void* d_out, int out_size, void* d_ws, size_t ws_size,
                              hipStream_t stream) {
    (void)in_sizes; (void)n_in; (void)out_size; (void)ws_size;
    const float* x     = (const float*)d_in[0];
    const float* W_ih0 = (const float*)d_in[1];
    const float* W_hh0 = (const float*)d_in[2];
    const float* b_ih0 = (const float*)d_in[3];
    const float* b_hh0 = (const float*)d_in[4];
    const float* W_ih1 = (const float*)d_in[5];
    const float* W_hh1 = (const float*)d_in[6];
    const float* b_ih1 = (const float*)d_in[7];
    const float* b_hh1 = (const float*)d_in[8];
    const float* fc_w  = (const float*)d_in[9];
    const float* fc_b  = (const float*)d_in[10];
    float* out = (float*)d_out;

    char* ws = (char*)d_ws;
    __bf16* xbf   = (__bf16*)(ws + OFF_XBF);
    float*  xw    = (float*) (ws + OFF_XW);
    __bf16* h1bf  = (__bf16*)(ws + OFF_H1);
    __bf16* pih0  = (__bf16*)(ws + OFF_WIH0);
    __bf16* phh0  = (__bf16*)(ws + OFF_WHH0);
    __bf16* pih1  = (__bf16*)(ws + OFF_WIH1);
    __bf16* phh1  = (__bf16*)(ws + OFF_WHH1);
    float*  bias0 = (float*) (ws + OFF_B0);
    float*  bias1 = (float*) (ws + OFF_B1);
    float*  hlast = (float*) (ws + OFF_HL);

    const int M = NT * NB;                                    // 131072 GEMM rows

    // phase 1: conversions / packing
    cvt_x_kernel<<<(size_t)M * NIN / 256, 256, 0, stream>>>(x, xbf);
    pack_w_kernel<<<(NH * NIN) / 256, 256, 0, stream>>>(W_ih0, pih0, NIN);
    pack_w_kernel<<<(NH * NH)  / 256, 256, 0, stream>>>(W_hh0, phh0, NH);
    pack_w_kernel<<<(NH * NH)  / 256, 256, 0, stream>>>(W_ih1, pih1, NH);
    pack_w_kernel<<<(NH * NH)  / 256, 256, 0, stream>>>(W_hh1, phh1, NH);
    add_bias_kernel<<<2, 256, 0, stream>>>(b_ih0, b_hh0, bias0);
    add_bias_kernel<<<2, 256, 0, stream>>>(b_ih1, b_hh1, bias1);

    const size_t scan_lds = (size_t)2 * NB * HSTR * sizeof(__bf16);  // 133 KB < 320 KB WGP LDS

    // layer 0: projection + scan (scan emits h1 in bf16 for layer-1 projection)
    gemm_bias_kernel<NIN><<<M / 32, 256, 0, stream>>>(xbf, pih0, bias0, xw, M);
    rnn_scan_kernel<<<1, 1024, scan_lds, stream>>>(xw, phh0, h1bf, nullptr, NT);

    // layer 1: projection (xw buffer reused) + scan (only final h needed)
    gemm_bias_kernel<NH><<<M / 32, 256, 0, stream>>>(h1bf, pih1, bias1, xw, M);
    rnn_scan_kernel<<<1, 1024, scan_lds, stream>>>(xw, phh1, nullptr, hlast, NT);

    // head: FC + softmax
    head_kernel<<<1, 64, 0, stream>>>(hlast, fc_w, fc_b, out);
}
